// CLIMATv1_15865609191962
// MI455X (gfx1250) — matrix operand order repaired and checked by reading, not compiled, not run on hardware
//
#include <hip/hip_runtime.h>
#include <hip/hip_bf16.h>

typedef __attribute__((ext_vector_type(16))) _Float16 v16h;
typedef __attribute__((ext_vector_type(8)))  _Float16 v8h;
typedef __attribute__((ext_vector_type(8)))  float    v8f;
typedef __attribute__((ext_vector_type(4)))  float    v4f;

#define DIMC   768
#define NTOK   257
#define BATCH  16
#define HEADS  12
#define HDIM   64
#define MLPD   3072
#define NCLS   1000
#define DEPTH  12
#define TOK    (BATCH * NTOK)   // 4112
#define ATT_SCALE 0.0360843918243516f   // 768^-0.5

#define SHUF16(lo, hi) __builtin_shufflevector((lo), (hi), 0,1,2,3,4,5,6,7,8,9,10,11,12,13,14,15)

// ---------------------------------------------------------------------------
// Weight pre-pack: f32 W[K][N]  ->  f16 Bp[(K/32)][N][32]
// Bp[((kblk*N)+n)*32 + kk] = W[kblk*32+kk][n]   (exact WMMA B-fragment image:
// a wave's 16 cols x 64B are contiguous -> coalesced global_load_b128 pairs)
// ---------------------------------------------------------------------------
__global__ __launch_bounds__(256) void pack_weight_f16(
    const float* __restrict__ W, _Float16* __restrict__ Bp, int K, int N)
{
    __shared__ float tile[32][33];
    const int n0 = blockIdx.x * 32;
    const int k0 = blockIdx.y * 32;          // kblk = blockIdx.y
    const int tx = threadIdx.x & 31, ty = threadIdx.x >> 5;
    for (int r = ty; r < 32; r += 8) {
        int gk = k0 + r, gn = n0 + tx;
        tile[r][tx] = (gk < K && gn < N) ? W[(size_t)gk * N + gn] : 0.f;
    }
    __syncthreads();
    const int u = threadIdx.x;
    if (u < 128) {
        int r = u >> 2, seg = (u & 3) * 8;   // r = n-within-tile, seg = k-chunk
        int gn = n0 + r;
        if (gn < N) {
            v8h hv;
#pragma unroll
            for (int j = 0; j < 8; j++) hv[j] = (_Float16)tile[seg + j][r];
            *(v8h*)&Bp[((size_t)blockIdx.y * N + gn) * 32 + seg] = hv;
        }
    }
}

// ---------------------------------------------------------------------------
// GEMM: C = act(A*Bp + bias) (+ residual). A:[M,K] f32; Bp: packed f16 weight.
// Block tile 128x128, BK=32, 8 waves, each wave 32x64 = 2x4 WMMA tiles.
// A staged in LDS (f32->f16); B fragments loaded DIRECTLY from packed global.
// ---------------------------------------------------------------------------
#define BM 128
#define BN 128
#define BK 32
#define LDA (BK + 8)     // 40 halves, 80B rows (16B aligned)

__global__ __launch_bounds__(256) void gemm_wmma_f16(
    const float* __restrict__ A, const _Float16* __restrict__ Bp,
    const float* __restrict__ bias, const float* __restrict__ residual,
    float* __restrict__ C, int M, int N, int K, int act_gelu)
{
    __shared__ _Float16 As[BM][LDA];

    const int tid   = threadIdx.x;
    const int lane  = tid & 31;
    const int wave  = tid >> 5;      // 0..7
    const int wm    = wave >> 1;     // 0..3  -> 32-row strip
    const int wn    = wave & 1;      // 0..1  -> 64-col strip
    const int bm0   = blockIdx.x * BM;
    const int bn0   = blockIdx.y * BN;
    const int l16   = lane & 15;
    const int lhalf = lane >> 4;

    v8f acc[2][4] = {};

    for (int k0 = 0; k0 < K; k0 += BK) {
        const int kblk = k0 >> 5;
        // ---- stage A: 128x32 halves as 512 runs-of-8; coalesced global rows
        for (int u = tid; u < BM * 4; u += 256) {
            int r = u >> 2, seg = (u & 3) * 8;
            int gr = bm0 + r;
            v8h hv = {};
            if (gr < M) {
                const float* g = &A[(size_t)gr * K + k0 + seg];
#pragma unroll
                for (int j = 0; j < 8; j++) hv[j] = (_Float16)g[j];
            }
            *(v8h*)&As[r][seg] = hv;
        }
        __syncthreads();

        // ---- prefetch next K-tile while math runs
        if (k0 + BK < K) {
            int pr = bm0 + (tid >> 1);
            if (pr < M)
                __builtin_prefetch(&A[(size_t)pr * K + k0 + BK + (tid & 1) * 16], 0, 1);
            int pc = bn0 + (tid >> 1);
            if (pc < N)
                __builtin_prefetch(&Bp[((size_t)(kblk + 1) * N + pc) * 32 + (tid & 1) * 16], 0, 1);
        }

        // ---- A fragments from LDS (contiguous b128 pairs)
        v16h afrag[2];
#pragma unroll
        for (int mt = 0; mt < 2; mt++) {
            int row = wm * 32 + mt * 16 + l16;
            v8h lo = *(const v8h*)&As[row][lhalf * 16];
            v8h hi = *(const v8h*)&As[row][lhalf * 16 + 8];
            afrag[mt] = SHUF16(lo, hi);
        }
        // ---- B fragments direct from packed global (coalesced b128 pairs)
        v16h bfrag[4];
#pragma unroll
        for (int nt = 0; nt < 4; nt++) {
            int col = bn0 + wn * 64 + nt * 16 + l16;
            v8h lo = {}, hi = {};
            if (col < N) {
                const _Float16* bp = &Bp[((size_t)kblk * N + col) * 32 + lhalf * 16];
                lo = *(const v8h*)bp;
                hi = *(const v8h*)(bp + 8);
            }
            bfrag[nt] = SHUF16(lo, hi);
        }
#pragma unroll
        for (int mt = 0; mt < 2; mt++)
#pragma unroll
            for (int nt = 0; nt < 4; nt++)
                acc[mt][nt] = __builtin_amdgcn_wmma_f32_16x16x32_f16(
                    false, afrag[mt], false, bfrag[nt], (short)0, acc[mt][nt],
                    false, false);
        __syncthreads();
    }

    // ---- epilogue
#pragma unroll
    for (int mt = 0; mt < 2; mt++) {
#pragma unroll
        for (int nt = 0; nt < 4; nt++) {
            int col = bn0 + wn * 64 + nt * 16 + l16;
            if (col >= N) continue;
#pragma unroll
            for (int r = 0; r < 8; r++) {
                int row = bm0 + wm * 32 + mt * 16 + r + 8 * lhalf;
                if (row >= M) continue;
                float v = acc[mt][nt][r];
                if (bias) v += bias[col];
                if (act_gelu) v = 0.5f * v * (1.f + erff(v * 0.70710678118654752f));
                if (residual) v += residual[(size_t)row * N + col];
                C[(size_t)row * N + col] = v;
            }
        }
    }
}

// ---------------------------------------------------------------------------
// Fused attention. Block = (16-query tile, head, batch), 128 threads.
// ---------------------------------------------------------------------------
#define ANT 17
#define ANP16 272
#define ANP32 288
#define KVELEMS (ANP16 * (HDIM + 8))      // >= 64*(272+8)

__global__ __launch_bounds__(128) void attn_wmma(
    const float* __restrict__ qkv, float* __restrict__ aout,
    float* __restrict__ attn_probs)
{
    __shared__ _Float16 kvraw[KVELEMS];
    __shared__ _Float16 Qs[16][HDIM + 8];
    __shared__ float    S[16][ANP32 + 8];
    __shared__ float    red[16][8];

    _Float16 (*Kb)[HDIM + 8]  = (_Float16 (*)[HDIM + 8])kvraw;   // [key][d]
    _Float16 (*Vt)[ANP16 + 8] = (_Float16 (*)[ANP16 + 8])kvraw;  // [d][key]

    const int mt = blockIdx.x, hI = blockIdx.y, b = blockIdx.z;
    const int tid   = threadIdx.x;
    const int lane  = tid & 31;
    const int wave  = tid >> 5;
    const int l16   = lane & 15;
    const int lhalf = lane >> 4;
    const size_t row0 = (size_t)b * NTOK;

    for (int u = tid; u < 16 * 8; u += 128) {
        int r = u >> 3, seg = (u & 7) * 8;
        int i = mt * 16 + r;
        v8h hv = {};
        if (i < NTOK) {
            const float* g = &qkv[(row0 + i) * 2304 + hI * HDIM + seg];
#pragma unroll
            for (int j = 0; j < 8; j++) hv[j] = (_Float16)g[j];
        }
        *(v8h*)&Qs[r][seg] = hv;
    }
    for (int u = tid; u < ANP16 * 8; u += 128) {
        int key = u >> 3, seg = (u & 7) * 8;
        v8h hv = {};
        if (key < NTOK) {
            const float* g = &qkv[(row0 + key) * 2304 + DIMC + hI * HDIM + seg];
#pragma unroll
            for (int j = 0; j < 8; j++) hv[j] = (_Float16)g[j];
        }
        *(v8h*)&Kb[key][seg] = hv;
    }
    __syncthreads();

    // scores: S = scale * Q @ K^T
    for (int nt = wave; nt < ANT; nt += 4) {
        v8f acc = {};
#pragma unroll
        for (int ks = 0; ks < 2; ks++) {
            v8h alo = *(const v8h*)&Qs[l16][ks * 32 + lhalf * 16];
            v8h ahi = *(const v8h*)&Qs[l16][ks * 32 + lhalf * 16 + 8];
            v16h af = SHUF16(alo, ahi);
            int key = nt * 16 + l16;
            v8h blo = *(const v8h*)&Kb[key][ks * 32 + lhalf * 16];
            v8h bhi = *(const v8h*)&Kb[key][ks * 32 + lhalf * 16 + 8];
            v16h bf = SHUF16(blo, bhi);
            acc = __builtin_amdgcn_wmma_f32_16x16x32_f16(false, af, false, bf,
                                                         (short)0, acc, false, false);
        }
#pragma unroll
        for (int r = 0; r < 8; r++)
            S[r + 8 * lhalf][nt * 16 + l16] = acc[r] * ATT_SCALE;
    }
    __syncthreads();

    // row softmax; zero padded cols; emit probabilities
    {
        int row = tid >> 3, sub = tid & 7;
        float mx = -1e30f;
        for (int j = sub; j < NTOK; j += 8) mx = fmaxf(mx, S[row][j]);
        red[row][sub] = mx;
        __syncthreads();
        float rmax = red[row][0];
#pragma unroll
        for (int s = 1; s < 8; s++) rmax = fmaxf(rmax, red[row][s]);
        __syncthreads();
        float sum = 0.f;
        for (int j = sub; j < ANP32; j += 8) {
            if (j < NTOK) { float e = __expf(S[row][j] - rmax); S[row][j] = e; sum += e; }
            else S[row][j] = 0.f;
        }
        red[row][sub] = sum;
        __syncthreads();
        float rsum = 0.f;
#pragma unroll
        for (int s = 0; s < 8; s++) rsum += red[row][s];
        float inv = 1.f / rsum;
        int qrow = mt * 16 + row;
        for (int j = sub; j < NTOK; j += 8) {
            float p = S[row][j] * inv;
            S[row][j] = p;
            if (qrow < NTOK)
                attn_probs[(((size_t)(b * HEADS + hI)) * NTOK + qrow) * NTOK + j] = p;
        }
    }
    __syncthreads();

    // stage V transposed into same buffer
    for (int u = tid; u < HDIM * (ANP16 / 8); u += 128) {
        int c = u & 63, seg = (u >> 6) * 8;
        v8h hv = {};
#pragma unroll
        for (int j = 0; j < 8; j++) {
            int key = seg + j;
            if (key < NTOK)
                hv[j] = (_Float16)qkv[(row0 + key) * 2304 + 2 * DIMC + hI * HDIM + c];
        }
        *(v8h*)&Vt[c][seg] = hv;
    }
    __syncthreads();

    // out = P @ V
    {
        int colbase = wave * 16;
        v8f acc = {};
#pragma unroll
        for (int kt = 0; kt < 9; kt++) {
            v4f s0 = *(const v4f*)&S[l16][kt * 32 + lhalf * 16];
            v4f s1 = *(const v4f*)&S[l16][kt * 32 + lhalf * 16 + 4];
            v4f s2 = *(const v4f*)&S[l16][kt * 32 + lhalf * 16 + 8];
            v4f s3 = *(const v4f*)&S[l16][kt * 32 + lhalf * 16 + 12];
            v16h af;
#pragma unroll
            for (int j = 0; j < 4; j++) {
                af[j]      = (_Float16)s0[j];
                af[4 + j]  = (_Float16)s1[j];
                af[8 + j]  = (_Float16)s2[j];
                af[12 + j] = (_Float16)s3[j];
            }
            int key0 = kt * 32 + lhalf * 16;
            v16h bf;
            if (key0 < ANP16) {
                v8h lo = *(const v8h*)&Vt[colbase + l16][key0];
                v8h hi = *(const v8h*)&Vt[colbase + l16][key0 + 8];
                bf = SHUF16(lo, hi);
            } else {
                v8h z = {};
                bf = SHUF16(z, z);
            }
            acc = __builtin_amdgcn_wmma_f32_16x16x32_f16(false, af, false, bf,
                                                         (short)0, acc, false, false);
        }
#pragma unroll
        for (int r = 0; r < 8; r++) {
            int qrow = mt * 16 + r + 8 * lhalf;
            if (qrow < NTOK)
                aout[((size_t)b * NTOK + qrow) * DIMC + hI * HDIM + colbase + l16] = acc[r];
        }
    }
}

// ---------------------------------------------------------------------------
__global__ __launch_bounds__(256) void ln_rows(
    const float* __restrict__ x, long row_stride,
    const float* __restrict__ g, const float* __restrict__ bb,
    float* __restrict__ out, int cols)
{
    __shared__ float sbuf[256];
    const float* xr = x + (size_t)blockIdx.x * row_stride;
    float* orow = out + (size_t)blockIdx.x * cols;
    int tid = threadIdx.x;
    float s = 0.f;
    for (int c = tid; c < cols; c += 256) s += xr[c];
    sbuf[tid] = s; __syncthreads();
    for (int off = 128; off > 0; off >>= 1) { if (tid < off) sbuf[tid] += sbuf[tid + off]; __syncthreads(); }
    float mu = sbuf[0] / cols;
    __syncthreads();
    float v = 0.f;
    for (int c = tid; c < cols; c += 256) { float d = xr[c] - mu; v += d * d; }
    sbuf[tid] = v; __syncthreads();
    for (int off = 128; off > 0; off >>= 1) { if (tid < off) sbuf[tid] += sbuf[tid + off]; __syncthreads(); }
    float inv = rsqrtf(sbuf[0] / cols + 1e-5f);
    for (int c = tid; c < cols; c += 256) orow[c] = (xr[c] - mu) * inv * g[c] + bb[c];
}

__global__ __launch_bounds__(256) void assemble_x0(
    const float* __restrict__ emb, const float* __restrict__ cls,
    const float* __restrict__ pos, float* __restrict__ x0)
{
    int t = blockIdx.x;
    int b = t / NTOK, i = t % NTOK;
    const float* src = (i == 0) ? cls : emb + (size_t)(b * 256 + i - 1) * DIMC;
    for (int c = threadIdx.x; c < DIMC; c += 256)
        x0[(size_t)t * DIMC + c] = src[c] + pos[(size_t)i * DIMC + c];
}

// ---------------------------------------------------------------------------
static inline dim3 gemm_grid(int M, int N) {
    return dim3((M + BM - 1) / BM, (N + BN - 1) / BN);
}
static inline dim3 pack_grid(int K, int N) {
    return dim3((N + 31) / 32, K / 32);
}

extern "C" void kernel_launch(void* const* d_in, const int* in_sizes, int n_in,
                              void* d_out, int out_size, void* d_ws, size_t ws_size,
                              hipStream_t stream) {
    (void)in_sizes; (void)n_in; (void)out_size; (void)ws_size;
    const float* features = (const float*)d_in[0];
    const float* W_embed  = (const float*)d_in[1];
    const float* b_embed  = (const float*)d_in[2];
    const float* cls_tok  = (const float*)d_in[3];
    const float* pos_emb  = (const float*)d_in[4];
    const float* ln0_g    = (const float*)d_in[5];
    const float* ln0_b    = (const float*)d_in[6];
    const float* Wqkv     = (const float*)d_in[7];
    const float* Wo       = (const float*)d_in[8];
    const float* bo       = (const float*)d_in[9];
    const float* ln1_g    = (const float*)d_in[10];
    const float* ln1_b    = (const float*)d_in[11];
    const float* W1       = (const float*)d_in[12];
    const float* b1       = (const float*)d_in[13];
    const float* W2       = (const float*)d_in[14];
    const float* b2       = (const float*)d_in[15];
    const float* hln_g    = (const float*)d_in[16];
    const float* hln_b    = (const float*)d_in[17];
    const float* hW1      = (const float*)d_in[18];
    const float* hb1      = (const float*)d_in[19];
    const float* hW2      = (const float*)d_in[20];
    const float* hb2      = (const float*)d_in[21];

    float* outp   = (float*)d_out;
    float* states = outp + (size_t)BATCH * NCLS;
    float* attns  = states + (size_t)DEPTH * TOK * DIMC;

    // ---- workspace: f32 buffers, then packed-f16 weight region
    float* ws = (float*)d_ws;
    size_t o = 0;
    float* emb    = ws + o; o += (size_t)BATCH * 256 * DIMC;
    float* x0     = ws + o; o += (size_t)TOK * DIMC;
    float* lnbuf  = ws + o; o += (size_t)TOK * DIMC;
    float* qkvbuf = ws + o; o += (size_t)TOK * 3 * DIMC;
    float* aoutb  = ws + o; o += (size_t)TOK * DIMC;
    float* mlpbuf = ws + o; o += (size_t)TOK * MLPD;
    float* hlnbuf = ws + o; o += (size_t)BATCH * DIMC;
    float* hmid   = ws + o; o += (size_t)BATCH * MLPD;

    _Float16* wpack = (_Float16*)(ws + o);
    _Float16* qkvP = wpack;                                    // 768*2304
    _Float16* woP  = qkvP + (size_t)DIMC * 3 * DIMC;           // 768*768
    _Float16* w1P  = woP  + (size_t)DIMC * DIMC;               // 768*3072
    _Float16* w2P  = w1P  + (size_t)DIMC * MLPD;               // 3072*768
    // embed / head packs alias the per-layer region (stream-ordered reuse)
    _Float16* embP  = wpack;
    _Float16* hW1P  = wpack;
    _Float16* hW2P  = wpack + (size_t)DIMC * MLPD;

    // ---- patch embedding
    pack_weight_f16<<<pack_grid(DIMC, DIMC), 256, 0, stream>>>(W_embed, embP, DIMC, DIMC);
    gemm_wmma_f16<<<gemm_grid(BATCH * 256, DIMC), 256, 0, stream>>>(
        features, embP, b_embed, nullptr, emb, BATCH * 256, DIMC, DIMC, 0);
    assemble_x0<<<TOK, 256, 0, stream>>>(emb, cls_tok, pos_emb, x0);

    for (int l = 0; l < DEPTH; l++) {
        const float* xin = (l == 0) ? x0 : states + (size_t)(l - 1) * TOK * DIMC;
        float* xres = states + (size_t)l * TOK * DIMC;
        float* aprob = attns + (size_t)l * BATCH * HEADS * NTOK * NTOK;

        pack_weight_f16<<<pack_grid(DIMC, 3 * DIMC), 256, 0, stream>>>(
            Wqkv + (size_t)l * DIMC * 3 * DIMC, qkvP, DIMC, 3 * DIMC);
        pack_weight_f16<<<pack_grid(DIMC, DIMC), 256, 0, stream>>>(
            Wo + (size_t)l * DIMC * DIMC, woP, DIMC, DIMC);
        pack_weight_f16<<<pack_grid(DIMC, MLPD), 256, 0, stream>>>(
            W1 + (size_t)l * DIMC * MLPD, w1P, DIMC, MLPD);
        pack_weight_f16<<<pack_grid(MLPD, DIMC), 256, 0, stream>>>(
            W2 + (size_t)l * MLPD * DIMC, w2P, MLPD, DIMC);

        ln_rows<<<TOK, 256, 0, stream>>>(xin, DIMC, ln0_g + l * DIMC,
                                         ln0_b + l * DIMC, lnbuf, DIMC);
        gemm_wmma_f16<<<gemm_grid(TOK, 3 * DIMC), 256, 0, stream>>>(
            lnbuf, qkvP, nullptr, nullptr, qkvbuf, TOK, 3 * DIMC, DIMC, 0);
        attn_wmma<<<dim3(ANT, HEADS, BATCH), 128, 0, stream>>>(qkvbuf, aoutb, aprob);
        gemm_wmma_f16<<<gemm_grid(TOK, DIMC), 256, 0, stream>>>(
            aoutb, woP, bo + l * DIMC, xin, xres, TOK, DIMC, DIMC, 0);
        ln_rows<<<TOK, 256, 0, stream>>>(xres, DIMC, ln1_g + l * DIMC,
                                         ln1_b + l * DIMC, lnbuf, DIMC);
        gemm_wmma_f16<<<gemm_grid(TOK, MLPD), 256, 0, stream>>>(
            lnbuf, w1P, b1 + l * MLPD, nullptr, mlpbuf, TOK, MLPD, DIMC, 1);
        gemm_wmma_f16<<<gemm_grid(TOK, DIMC), 256, 0, stream>>>(
            mlpbuf, w2P, b2 + l * DIMC, xres, xres, TOK, DIMC, MLPD, 0);
    }

    // ---- head
    ln_rows<<<BATCH, 256, 0, stream>>>(states + (size_t)(DEPTH - 1) * TOK * DIMC,
                                       (long)NTOK * DIMC, hln_g, hln_b, hlnbuf, DIMC);
    pack_weight_f16<<<pack_grid(DIMC, MLPD), 256, 0, stream>>>(hW1, hW1P, DIMC, MLPD);
    gemm_wmma_f16<<<gemm_grid(BATCH, MLPD), 256, 0, stream>>>(
        hlnbuf, hW1P, hb1, nullptr, hmid, BATCH, MLPD, DIMC, 1);
    pack_weight_f16<<<pack_grid(MLPD, NCLS), 256, 0, stream>>>(hW2, hW2P, MLPD, NCLS);
    gemm_wmma_f16<<<gemm_grid(BATCH, NCLS), 256, 0, stream>>>(
        hmid, hW2P, hb2, nullptr, outp, BATCH, NCLS, MLPD, 0);
}
